// LSTMAutoregressive_2731599200267
// MI455X (gfx1250) — matrix operand order repaired and checked by reading, not compiled, not run on hardware
//
#include <hip/hip_runtime.h>
#include <hip/hip_bf16.h>

#define B_   4096
#define S_   1024
#define I_   6
#define H_   128
#define P_   50
#define BT   16            // batch rows per workgroup
#define NWG  (B_ / BT)     // 256 workgroups
#define NTHR 256           // 8 wave32s

// padded LDS strides (in halves): 16B-aligned rows with bank rotation
#define LDX  40            // xbuf row stride   (80B  = 20 words -> rotate 20 banks/row)
#define LDH  136           // h-state row stride(272B = 68 words -> rotate 4 banks/row)
#define LDW0 136           // LDS whh0 row stride
#define LDI0 40            // LDS wih0 row stride (80B, conflict-free rotation)

typedef __attribute__((ext_vector_type(8)))  _Float16 v8h;
typedef __attribute__((ext_vector_type(16))) _Float16 v16h;
typedef __attribute__((ext_vector_type(8)))  float    v8f;

// ---- workspace layout (byte offsets) ----
#define WS_WIH0 0          // 512x32  f16 (cols 0..5 = w_ih0, col 6 = b0, rest 0)
#define WS_WHH0 32768      // 512x128 f16
#define WS_WIH1 163840     // 512x128 f16
#define WS_WHH1 294912     // 512x128 f16
#define WS_FCW  425984     // 16x128  f16 (padded from 6x128)
#define WS_B1   430080     // 512 f32 (b_ih1+b_hh1)
#define WS_FCB  432128     // 16  f32 (padded)
#define PREP_TOTAL 215568

__device__ __forceinline__ float sigf(float x) { return 1.0f / (1.0f + __expf(-x)); }
__device__ __forceinline__ float tanhfast(float x) {
  float e = __expf(-2.0f * __builtin_fabsf(x));
  float y = (1.0f - e) / (1.0f + e);
  return __builtin_copysignf(y, x);
}

// A fragment, 16x32 f16: lane m=L&15; lanes<16 hold K {0..15},
// lanes>=16 hold K {16..31} -> two adjacent 16B chunks per lane.
__device__ __forceinline__ v16h ld_a(const _Float16* base, int k0, int ld, int lane) {
  int row = lane & 15;
  int kb  = k0 + (lane & 16);                  // 0 or 16
  const _Float16* p = base + row * ld + kb;
  v8h lo = *(const v8h*)(p);
  v8h hi = *(const v8h*)(p + 8);
  return __builtin_shufflevector(lo, hi, 0,1,2,3,4,5,6,7,8,9,10,11,12,13,14,15);
}

// B fragment, 32x16 f16: lane n=L&15 holds column n = weight row (n0+n),
// K = [k0 .. k0+15] for lanes<16, [k0+16 .. k0+31] for lanes>=16: one 32B run.
__device__ __forceinline__ v16h ld_b(const _Float16* W, int n0, int k0, int ld, int lane) {
  int row = n0 + (lane & 15);
  int kb  = k0 + (lane & 16);
  return *(const v16h*)(W + row * ld + kb);
}

// B fragment from a shared LDS copy: plain ds_load_b128 pair. The caller mixes
// in a laundered (opaque) zero offset so these cannot be hoisted/CSE'd out of
// the time loop (which would re-create register spilling).
__device__ __forceinline__ v16h ld_b_lds(const _Float16* W, int off) {
  const v8h* p = (const v8h*)(W + off);
  v8h lo = p[0];
  v8h hi = p[1];
  return __builtin_shufflevector(lo, hi, 0,1,2,3,4,5,6,7,8,9,10,11,12,13,14,15);
}

// Layer 0: both ih and hh fragments streamed from shared LDS weight copies.
// Bias folded into input column 6.
__device__ __forceinline__ void lstm_layer0(
    const _Float16* __restrict__ xb,
    const _Float16* __restrict__ hin,
    _Float16* __restrict__ hout,
    const _Float16* wih0s,
    const _Float16* whh0s,
    v8f& cst, int lane, int cb)
{
  // opaque zero, redefined every call (i.e., every loop iteration): defeats
  // LICM/GVN on the LDS weight-fragment loads below while keeping them ds_load.
  int wofs = 0;
  asm volatile("" : "+v"(wofs));

  v8f acc[4];
#pragma unroll
  for (int g = 0; g < 4; ++g) acc[g] = (v8f){0,0,0,0,0,0,0,0};

  v16h ax = ld_a(xb, 0, LDX, lane);
#pragma unroll
  for (int g = 0; g < 4; ++g) {
    int off = (g * H_ + cb + (lane & 15)) * LDI0 + (lane & 16) + wofs;
    v16h b = ld_b_lds(wih0s, off);
    acc[g] = __builtin_amdgcn_wmma_f32_16x16x32_f16(false, ax, false, b,
                                                    (short)0, acc[g], false, false);
  }
#pragma unroll
  for (int kt = 0; kt < 4; ++kt) {
    v16h a = ld_a(hin, kt * 32, LDH, lane);
#pragma unroll
    for (int g = 0; g < 4; ++g) {
      int off = (g * H_ + cb + (lane & 15)) * LDW0 + kt * 32 + (lane & 16) + wofs;
      v16h b = ld_b_lds(whh0s, off);
      acc[g] = __builtin_amdgcn_wmma_f32_16x16x32_f16(false, a, false, b,
                                                      (short)0, acc[g], false, false);
    }
  }

  int coff = cb + (lane & 15);
  int radd = (lane & 16) ? 8 : 0;
#pragma unroll
  for (int r = 0; r < 8; ++r) {
    float cc = sigf(acc[1][r]) * cst[r] + sigf(acc[0][r]) * tanhfast(acc[2][r]);
    cst[r] = cc;
    hout[(r + radd) * LDH + coff] = (_Float16)(sigf(acc[3][r]) * tanhfast(cc));
  }
  __syncthreads();
}

// Layer 1: all 32 weight fragments register-resident; biases read from a small
// LDS array at gate-update time (laundered, so they never occupy loop-live VGPRs).
__device__ __forceinline__ void lstm_layer1(
    const _Float16* __restrict__ inp,
    const _Float16* __restrict__ hin,
    _Float16* __restrict__ hout,
    const v16h (&w)[4][8],
    const float* b1s,
    v8f& cst, int lane, int cb)
{
  int bofs = 0;
  asm volatile("" : "+v"(bofs));

  v8f acc[4];
#pragma unroll
  for (int g = 0; g < 4; ++g) acc[g] = (v8f){0,0,0,0,0,0,0,0};

#pragma unroll
  for (int kt = 0; kt < 4; ++kt) {             // input (h0) contribution
    v16h a = ld_a(inp, kt * 32, LDH, lane);
#pragma unroll
    for (int g = 0; g < 4; ++g)
      acc[g] = __builtin_amdgcn_wmma_f32_16x16x32_f16(false, a, false, w[g][kt],
                                                      (short)0, acc[g], false, false);
  }
#pragma unroll
  for (int kt = 0; kt < 4; ++kt) {             // recurrent (h1) contribution
    v16h a = ld_a(hin, kt * 32, LDH, lane);
#pragma unroll
    for (int g = 0; g < 4; ++g)
      acc[g] = __builtin_amdgcn_wmma_f32_16x16x32_f16(false, a, false, w[g][4 + kt],
                                                      (short)0, acc[g], false, false);
  }

  int coff = cb + (lane & 15);
  float bi = b1s[bofs + coff];
  float bf = b1s[bofs + H_ + coff];
  float bg = b1s[bofs + 2 * H_ + coff];
  float bo = b1s[bofs + 3 * H_ + coff];
  int radd = (lane & 16) ? 8 : 0;
#pragma unroll
  for (int r = 0; r < 8; ++r) {
    float iv = acc[0][r] + bi, fv = acc[1][r] + bf;
    float gv = acc[2][r] + bg, ov = acc[3][r] + bo;
    float cc = sigf(fv) * cst[r] + sigf(iv) * tanhfast(gv);
    cst[r] = cc;
    hout[(r + radd) * LDH + coff] = (_Float16)(sigf(ov) * tanhfast(cc));
  }
  __syncthreads();
}

// pred = tanh(h1 @ fc_w^T + fc_b) -> xbuf (fp16, next AR input) and optionally out.
__device__ __forceinline__ void fc_stage(const _Float16* __restrict__ h1s,
    const _Float16* __restrict__ fcw, const float* __restrict__ fcbv,
    _Float16* __restrict__ xbuf, float* __restrict__ out,
    int wb, int p, bool emit, int lane, int wave)
{
  if (wave == 0) {
    float fcb_l = fcbv[lane & 15];
    v8f acc = (v8f){0,0,0,0,0,0,0,0};
#pragma unroll
    for (int kt = 0; kt < 4; ++kt) {
      v16h a = ld_a(h1s, kt * 32, LDH, lane);
      v16h b = ld_b(fcw, 0, kt * 32, H_, lane);
      acc = __builtin_amdgcn_wmma_f32_16x16x32_f16(false, a, false, b,
                                                   (short)0, acc, false, false);
    }
    int col  = lane & 15;
    int radd = (lane & 16) ? 8 : 0;
#pragma unroll
    for (int r = 0; r < 8; ++r) {
      int m = r + radd;
      float v = tanhfast(acc[r] + fcb_l);
      if (col < I_) {
        xbuf[m * LDX + col] = (_Float16)v;
        if (emit) out[((size_t)(wb + m) * P_ + p) * I_ + col] = v;
      }
    }
  }
  __syncthreads();
}

__global__ __launch_bounds__(NTHR, 2) void lstm_ar_kernel(
    const float* __restrict__ x,
    const _Float16* __restrict__ wih0, const _Float16* __restrict__ whh0,
    const _Float16* __restrict__ wih1, const _Float16* __restrict__ whh1,
    const _Float16* __restrict__ fcw,
    const float* __restrict__ b1v, const float* __restrict__ fcbv,
    float* __restrict__ out)
{
  __shared__ alignas(32) _Float16 whh0s[512 * LDW0];  // shared L0 recurrent weights
  __shared__ alignas(32) _Float16 wih0s[512 * LDI0];  // shared L0 input weights (+bias col)
  __shared__ alignas(32) _Float16 h0s[2][BT * LDH];   // ping-pong hidden states
  __shared__ alignas(32) _Float16 h1s[2][BT * LDH];
  __shared__ alignas(32) _Float16 xbuf[BT * LDX];     // layer-0 input (col6 == 1.0)
  __shared__ alignas(16) float    b1s[4 * H_];        // layer-1 gate biases

  int tid  = threadIdx.x;
  int lane = tid & 31;
  int wave = tid >> 5;
  int cb   = wave * 16;          // this wave's gate-column block
  int wb   = blockIdx.x * BT;    // this WG's batch rows

  // stage weights into padded LDS copies (32-bit copies, one time)
  {
    const uint32_t* src = (const uint32_t*)whh0;      // 512 rows x 64 dwords
    uint32_t* dst = (uint32_t*)whh0s;                 // row stride 68 dwords
    for (int i = tid; i < 512 * 64; i += NTHR) {
      int r = i >> 6, c = i & 63;
      dst[r * (LDW0 / 2) + c] = src[i];
    }
    const uint32_t* s2 = (const uint32_t*)wih0;       // 512 rows x 16 dwords
    uint32_t* d2 = (uint32_t*)wih0s;                  // row stride 20 dwords
    for (int i = tid; i < 512 * 16; i += NTHR) {
      int r = i >> 4, c = i & 15;
      d2[r * (LDI0 / 2) + c] = s2[i];
    }
    for (int i = tid; i < 4 * H_; i += NTHR) b1s[i] = b1v[i];
  }
  for (int i = tid; i < BT * LDX; i += NTHR) xbuf[i] = (_Float16)((i % LDX) == I_ ? 1.0f : 0.0f);
  for (int i = tid; i < 2 * BT * LDH; i += NTHR) { h0s[0][i] = (_Float16)0.0f; h1s[0][i] = (_Float16)0.0f; }
  __syncthreads();

  // ---- register-resident weights: full L1 set only (256 VGPRs) ----
  v16h w1[4][8];   // [gate][kt]: kt0..3 = ih (h0), kt4..7 = hh (h1)
#pragma unroll
  for (int g = 0; g < 4; ++g) {
    int n0 = g * H_ + cb;
#pragma unroll
    for (int kt = 0; kt < 4; ++kt) {
      w1[g][kt]     = ld_b(wih1, n0, kt * 32, H_, lane);
      w1[g][4 + kt] = ld_b(whh1, n0, kt * 32, H_, lane);
    }
  }

  v8f c0 = (v8f){0,0,0,0,0,0,0,0};
  v8f c1 = (v8f){0,0,0,0,0,0,0,0};

  // ---------------- encode: S sequential steps ----------------
  for (int t = 0; t < S_; ++t) {
    int pr = t & 1, pw = pr ^ 1;
    if (tid < BT * I_) {                       // stage x_t tile (fp32 -> fp16)
      int rr = tid / I_, ii = tid - rr * I_;
      xbuf[rr * LDX + ii] = (_Float16)x[((size_t)(wb + rr) * S_ + t) * I_ + ii];
    }
    __syncthreads();
    lstm_layer0(xbuf, h0s[pr], h0s[pw], wih0s, whh0s, c0, lane, cb);
    lstm_layer1(h0s[pw], h1s[pr], h1s[pw], w1, b1s, c1, lane, cb);
  }
  // after S_ (even) steps, live state is in buffer index 0

  // seed cur = tanh(fc(h1)) (not emitted)
  fc_stage(h1s[0], fcw, fcbv, xbuf, out, wb, 0, false, lane, wave);

  // ---------------- autoregressive: P steps ----------------
  for (int p = 0; p < P_; ++p) {
    int pr = p & 1, pw = pr ^ 1;
    lstm_layer0(xbuf, h0s[pr], h0s[pw], wih0s, whh0s, c0, lane, cb);
    lstm_layer1(h0s[pw], h1s[pr], h1s[pw], w1, b1s, c1, lane, cb);
    fc_stage(h1s[pw], fcw, fcbv, xbuf, out, wb, p, true, lane, wave);
  }
}

// ---- prologue: fp32 -> fp16 weight conversion + padding + bias folding ----
__global__ void prep_kernel(
    const float* __restrict__ w_ih0, const float* __restrict__ w_hh0,
    const float* __restrict__ b_ih0, const float* __restrict__ b_hh0,
    const float* __restrict__ w_ih1, const float* __restrict__ w_hh1,
    const float* __restrict__ b_ih1, const float* __restrict__ b_hh1,
    const float* __restrict__ fc_w,  const float* __restrict__ fc_b,
    _Float16* wih0h, _Float16* whh0h, _Float16* wih1h, _Float16* whh1h,
    _Float16* fcwh, float* b1, float* fcb)
{
  int idx = blockIdx.x * blockDim.x + threadIdx.x;
  if (idx < 16384) {                               // w_ih0: 512x6 -> 512x32; col6 = b0
    int r = idx >> 5, c = idx & 31;
    float v = (c < I_) ? w_ih0[r * I_ + c]
            : (c == I_) ? (b_ih0[r] + b_hh0[r]) : 0.0f;
    wih0h[idx] = (_Float16)v;
  } else if (idx < 81920) {                        // w_hh0 512x128
    int j = idx - 16384;  whh0h[j] = (_Float16)w_hh0[j];
  } else if (idx < 147456) {                       // w_ih1 512x128
    int j = idx - 81920;  wih1h[j] = (_Float16)w_ih1[j];
  } else if (idx < 212992) {                       // w_hh1 512x128
    int j = idx - 147456; whh1h[j] = (_Float16)w_hh1[j];
  } else if (idx < 215040) {                       // fc_w: 6x128 -> 16x128 padded
    int j = idx - 212992; int r = j >> 7, c = j & 127;
    fcwh[j] = (_Float16)(r < I_ ? fc_w[r * H_ + c] : 0.0f);
  } else if (idx < 215552) {                       // b1 = b_ih1 + b_hh1
    int j = idx - 215040; b1[j] = b_ih1[j] + b_hh1[j];
  } else if (idx < PREP_TOTAL) {                   // fc_b padded to 16
    int j = idx - 215552; fcb[j] = (j < I_) ? fc_b[j] : 0.0f;
  }
}

extern "C" void kernel_launch(void* const* d_in, const int* in_sizes, int n_in,
                              void* d_out, int out_size, void* d_ws, size_t ws_size,
                              hipStream_t stream) {
  (void)in_sizes; (void)n_in; (void)out_size; (void)ws_size;
  const float* x     = (const float*)d_in[0];
  const float* w_ih0 = (const float*)d_in[1];
  const float* w_hh0 = (const float*)d_in[2];
  const float* b_ih0 = (const float*)d_in[3];
  const float* b_hh0 = (const float*)d_in[4];
  const float* w_ih1 = (const float*)d_in[5];
  const float* w_hh1 = (const float*)d_in[6];
  const float* b_ih1 = (const float*)d_in[7];
  const float* b_hh1 = (const float*)d_in[8];
  const float* fc_w  = (const float*)d_in[9];
  const float* fc_b  = (const float*)d_in[10];
  float* out = (float*)d_out;

  char* ws = (char*)d_ws;
  _Float16* wih0h = (_Float16*)(ws + WS_WIH0);
  _Float16* whh0h = (_Float16*)(ws + WS_WHH0);
  _Float16* wih1h = (_Float16*)(ws + WS_WIH1);
  _Float16* whh1h = (_Float16*)(ws + WS_WHH1);
  _Float16* fcwh  = (_Float16*)(ws + WS_FCW);
  float* b1  = (float*)(ws + WS_B1);
  float* fcb = (float*)(ws + WS_FCB);

  prep_kernel<<<(PREP_TOTAL + 255) / 256, 256, 0, stream>>>(
      w_ih0, w_hh0, b_ih0, b_hh0, w_ih1, w_hh1, b_ih1, b_hh1, fc_w, fc_b,
      wih0h, whh0h, wih1h, whh1h, fcwh, b1, fcb);

  lstm_ar_kernel<<<NWG, NTHR, 0, stream>>>(
      x, wih0h, whh0h, wih1h, whh1h, fcwh, b1, fcb, out);
}
